// BilstmCrf_89386859364781
// MI455X (gfx1250) — matrix operand order repaired and checked by reading, not compiled, not run on hardware
//
#include <hip/hip_runtime.h>
#include <hip/hip_bf16.h>

// ---------------- types / constants ----------------
typedef __bf16 bf16_t;
typedef __attribute__((ext_vector_type(16))) __bf16 bf16x16;
typedef __attribute__((ext_vector_type(8)))  __bf16 bf16x8;
typedef __attribute__((ext_vector_type(8)))  float  f32x8;

#define D_IN   300
#define KPAD   320      // D padded to multiple of 32 for WMMA K
#define HID    256
#define G4H    1024     // 4*H gate width
#define NTAG   34
#define BATCH  64
#define SEQ    256
#define MROWS  (SEQ*BATCH)   // 16384 flattened (s,b) rows

__device__ __forceinline__ float sigmoidf_(float x) { return 1.0f / (1.0f + __expf(-x)); }

// ---------------- WMMA helpers (gfx1250 bf16 16x16x32, fp32 accum) ----------------
__device__ __forceinline__ f32x8 wmma_bf16(bf16x16 a, bf16x16 b, f32x8 c) {
  return __builtin_amdgcn_wmma_f32_16x16x32_bf16(false, a, false, b, (short)0, c, false, false);
}

// A fragment: 16x32 bf16 tile from a row-major [rows][stride] bf16 buffer.
// ISA layout: lane L holds row m = L%16; elems 0..7 = K run at kbase+(L/16)*8,
// elems 8..15 = K run at kbase+16+(L/16)*8 (two contiguous 16B loads).
__device__ __forceinline__ bf16x16 load_a_bf16(const bf16_t* base, int stride,
                                               int row0, int lane, int kbase) {
  int m  = row0 + (lane & 15);
  int hi = lane >> 4;
  const bf16_t* p = base + (size_t)m * stride + kbase + hi * 8;
  bf16x8 lo = *(const bf16x8*)p;
  bf16x8 hi8 = *(const bf16x8*)(p + 16);
  bf16x16 a;
#pragma unroll
  for (int i = 0; i < 8; ++i) { a[i] = lo[i]; a[8 + i] = hi8[i]; }
  return a;
}

// Same layout but source is fp32 (convert on the fly).
__device__ __forceinline__ bf16x16 load_a_f32(const float* base, int stride,
                                              int row0, int lane, int kbase) {
  int m  = row0 + (lane & 15);
  int hi = lane >> 4;
  const float* p = base + (size_t)m * stride + kbase + hi * 8;
  bf16x16 a;
#pragma unroll
  for (int i = 0; i < 8; ++i) { a[i] = (bf16_t)p[i]; a[8 + i] = (bf16_t)p[16 + i]; }
  return a;
}

// B fragments are PREPACKED so each lane just loads 32 contiguous bytes:
// packedB[(kk*NT + nt)*32*16 + lane*16 + j]  where j elem -> K = (j/8)*16 + (lane/16)*8 + j%8,
// column n = nt*16 + lane%16.
__device__ __forceinline__ bf16x16 load_b_packed(const bf16_t* packed, int kk, int NT,
                                                 int nt, int lane) {
  return *(const bf16x16*)(packed + (((size_t)kk * NT + nt) * 32 + lane) * 16);
}

// ---------------- prep kernels: gather/pack to bf16 ----------------
// X[r=s*64+b][0..319] = emb[batch[b][s]][d] (zero pad d>=300)
__global__ void pack_x_kernel(const int* __restrict__ batch, const float* __restrict__ emb,
                              bf16_t* __restrict__ X) {
  int idx = blockIdx.x * 256 + threadIdx.x;           // MROWS*KPAD threads
  int r = idx / KPAD, d = idx % KPAD;
  int s = r >> 6, b = r & 63;
  int tok = batch[b * SEQ + s];
  float v = (d < D_IN) ? emb[(size_t)tok * D_IN + d] : 0.0f;
  X[idx] = (bf16_t)v;
}

// w_ih (both dirs) -> WMMA-B layout, K padded 300->320. [dir][kt(10)][nt(64)][lane(32)][j(16)]
__global__ void pack_wih_kernel(const float* __restrict__ wf, const float* __restrict__ wb,
                                bf16_t* __restrict__ out) {
  int idx = blockIdx.x * 256 + threadIdx.x;           // 2*10*64*512
  int j = idx & 15, l = (idx >> 4) & 31, nt = (idx >> 9) & 63;
  int kt = (idx >> 15) % 10, dir = idx / (10 << 15);
  int n = nt * 16 + (l & 15);
  int k = kt * 32 + ((j >> 3) << 4) + ((l >> 4) << 3) + (j & 7);
  const float* src = dir ? wb : wf;
  float v = (k < D_IN) ? src[(size_t)n * D_IN + k] : 0.0f;
  out[idx] = (bf16_t)v;
}

// w_hh -> B layout, K=256. [dir][kt(8)][nt(64)][32][16]
__global__ void pack_whh_kernel(const float* __restrict__ wf, const float* __restrict__ wb,
                                bf16_t* __restrict__ out) {
  int idx = blockIdx.x * 256 + threadIdx.x;           // 2*8*64*512 = 2^19
  int j = idx & 15, l = (idx >> 4) & 31, nt = (idx >> 9) & 63;
  int kt = (idx >> 15) & 7, dir = idx >> 18;
  int n = nt * 16 + (l & 15);
  int k = kt * 32 + ((j >> 3) << 4) + ((l >> 4) << 3) + (j & 7);
  const float* src = dir ? wb : wf;
  out[idx] = (bf16_t)src[(size_t)n * HID + k];
}

// w_out [34][512] -> B layout, N padded 34->48. [kt(16)][nt(3)][32][16]
__global__ void pack_wout_kernel(const float* __restrict__ w, bf16_t* __restrict__ out) {
  int idx = blockIdx.x * 256 + threadIdx.x;           // 16*3*512
  int j = idx & 15, l = (idx >> 4) & 31;
  int nt = (idx >> 9) % 3, kt = idx / (3 << 9);
  int t = nt * 16 + (l & 15);
  int k = kt * 32 + ((j >> 3) << 4) + ((l >> 4) << 3) + (j & 7);
  float v = (t < NTAG) ? w[(size_t)t * (2 * HID) + k] : 0.0f;
  out[idx] = (bf16_t)v;
}

// ---------------- input projection GEMM: xg[dir][s][b][1024] ----------------
// 2 dirs x 1024 Mtiles x 64 Ntiles; one 16x16 tile per wave, K loop 10x32.
__global__ void xproj_gemm_kernel(const bf16_t* __restrict__ X, const bf16_t* __restrict__ wihB,
                                  const float* __restrict__ bias_f, const float* __restrict__ bias_b,
                                  float* __restrict__ xg) {
  int wave = threadIdx.x >> 5, lane = threadIdx.x & 31;
  int tile = blockIdx.x * 8 + wave;
  int nt = tile & 63, mt = (tile >> 6) & 1023, dir = tile >> 16;
  const bf16_t* wB = wihB + (size_t)dir * 10 * 64 * 512;
  f32x8 acc = {0.f, 0.f, 0.f, 0.f, 0.f, 0.f, 0.f, 0.f};
  for (int kk = 0; kk < 10; ++kk) {
    bf16x16 a = load_a_bf16(X, KPAD, mt * 16, lane, kk * 32);
    bf16x16 b = load_b_packed(wB, kk, 64, nt, lane);
    acc = wmma_bf16(a, b, acc);
  }
  const float* bias = dir ? bias_b : bias_f;
  int n = nt * 16 + (lane & 15);
  float bv = bias[n];
  int hi = lane >> 4;
#pragma unroll
  for (int v = 0; v < 8; ++v) {
    int r = mt * 16 + hi * 8 + v;
    xg[((size_t)dir * MROWS + r) * G4H + n] = acc[v] + bv;
  }
}

// ---------------- persistent LSTM scan: 1 workgroup per direction ----------------
// h-state (bf16, WMMA A operand) lives in LDS; c-state lives in registers.
// Per step: 64x1024 GEMM (K=256) tiled 4x64, 32 waves x (2 combos x 4 gates).
__global__ void lstm_scan_kernel(const float* __restrict__ xg, const bf16_t* __restrict__ whhB,
                                 float* __restrict__ hidden) {
  const int dir  = blockIdx.x;
  const int tid  = threadIdx.x;
  const int wave = tid >> 5, lane = tid & 31;
  const int hi = lane >> 4, ln = lane & 15;

  __shared__ bf16_t hbuf[BATCH * HID];                // 32 KB of the 320 KB WGP LDS
  for (int i = tid; i < BATCH * HID; i += 1024) hbuf[i] = (bf16_t)0.0f;
  __syncthreads();

  const bf16_t* wB  = whhB + (size_t)dir * 8 * 64 * 512;
  const float*  xgd = xg + (size_t)dir * MROWS * G4H;

  f32x8 cstate[2] = {};                               // persistent cell state
  const int combo0 = wave * 2;                        // combo = mt*16 + t  (mt:0..3, t:0..15)

  for (int si = 0; si < SEQ; ++si) {
    const int s = dir ? (SEQ - 1 - si) : si;
    __builtin_prefetch(xgd + (size_t)s * BATCH * G4H + tid * 64, 0, 1);

    f32x8 acc[2][4] = {};
#pragma unroll
    for (int c = 0; c < 2; ++c) {
      const int combo = combo0 + c;
      const int mt = combo >> 4, t = combo & 15;
#pragma unroll
      for (int kk = 0; kk < 8; ++kk) {
        bf16x16 a = load_a_bf16(hbuf, HID, mt * 16, lane, kk * 32);
#pragma unroll
        for (int g = 0; g < 4; ++g)                    // N-tile = g*16 + t (gate-major layout)
          acc[c][g] = wmma_bf16(a, load_b_packed(wB, kk, 64, g * 16 + t, lane), acc[c][g]);
      }
    }
    __syncthreads();                                   // all reads of hbuf done (WAR)

    const float* xrow = xgd + (size_t)s * BATCH * G4H;
#pragma unroll
    for (int c = 0; c < 2; ++c) {
      const int combo = combo0 + c;
      const int mt = combo >> 4, t = combo & 15;
      const int n = t * 16 + ln;                       // 0..255 within H
#pragma unroll
      for (int v = 0; v < 8; ++v) {
        const int b = mt * 16 + hi * 8 + v;
        const float* xb = xrow + (size_t)b * G4H + n;
        const float gi = acc[c][0][v] + xb[0];
        const float gf = acc[c][1][v] + xb[256];
        const float gg = acc[c][2][v] + xb[512];
        const float go = acc[c][3][v] + xb[768];
        const float cn = sigmoidf_(gf) * cstate[c][v] + sigmoidf_(gi) * tanhf(gg);
        cstate[c][v] = cn;
        const float h = sigmoidf_(go) * tanhf(cn);
        hbuf[b * HID + n] = (bf16_t)h;
        hidden[((size_t)s * BATCH + b) * (2 * HID) + dir * HID + n] = h;
      }
    }
    __syncthreads();                                   // hbuf writes visible (RAW)
  }
}

// ---------------- output projection: emissions[b][s][t] ----------------
__global__ void outproj_kernel(const float* __restrict__ hidden, const bf16_t* __restrict__ woutB,
                               const float* __restrict__ b_out, float* __restrict__ emis) {
  int wave = threadIdx.x >> 5, lane = threadIdx.x & 31;
  int tile = blockIdx.x * 8 + wave;                    // 1024 mt x 3 nt
  int nt = tile % 3, mt = tile / 3;
  f32x8 acc = {0.f, 0.f, 0.f, 0.f, 0.f, 0.f, 0.f, 0.f};
  for (int kk = 0; kk < 16; ++kk) {
    bf16x16 a = load_a_f32(hidden, 2 * HID, mt * 16, lane, kk * 32);
    bf16x16 b = load_b_packed(woutB, kk, 3, nt, lane);
    acc = wmma_bf16(a, b, acc);
  }
  int t = nt * 16 + (lane & 15);
  if (t < NTAG) {
    float bo = b_out[t];
    int hi = lane >> 4;
#pragma unroll
    for (int v = 0; v < 8; ++v) {
      int r = mt * 16 + hi * 8 + v;
      int s = r >> 6, b = r & 63;
      emis[((size_t)b * SEQ + s) * NTAG + t] = acc[v] + bo;
    }
  }
}

// ---------------- CRF negative log-likelihood: 1 block per sequence ----------------
__global__ void crf_kernel(const float* __restrict__ emis, const int* __restrict__ tags,
                           const float* __restrict__ start_t, const float* __restrict__ end_t,
                           const float* __restrict__ trans, float* __restrict__ llh) {
  const int b = blockIdx.x, tid = threadIdx.x;         // 64 threads
  __shared__ float sT[NTAG * NTAG];
  __shared__ float buf0[NTAG], buf1[NTAG];
  __shared__ float red[64];
  for (int i = tid; i < NTAG * NTAG; i += 64) sT[i] = trans[i];
  const float* em = emis + (size_t)b * SEQ * NTAG;
  if (tid < NTAG) buf0[tid] = start_t[tid] + em[tid];
  __syncthreads();

  float* cur = buf0; float* nxt = buf1;
  for (int s = 1; s < SEQ; ++s) {
    if (tid < NTAG) {
      float m = -3.0e38f;
      for (int k = 0; k < NTAG; ++k) m = fmaxf(m, cur[k] + sT[k * NTAG + tid]);
      float sum = 0.0f;
      for (int k = 0; k < NTAG; ++k) sum += __expf(cur[k] + sT[k * NTAG + tid] - m);
      nxt[tid] = m + __logf(sum) + em[s * NTAG + tid];
    }
    __syncthreads();
    float* tmp = cur; cur = nxt; nxt = tmp;
  }

  // gold-path numerator (parallel over s, tree reduce)
  const int* tg = tags + b * SEQ;
  float p = 0.0f;
  for (int s = tid; s < SEQ; s += 64) {
    int t = tg[s];
    p += em[s * NTAG + t];
    if (s > 0) p += sT[tg[s - 1] * NTAG + t];
  }
  red[tid] = p;
  __syncthreads();
  for (int off = 32; off > 0; off >>= 1) {
    if (tid < off) red[tid] += red[tid + off];
    __syncthreads();
  }
  if (tid == 0) {
    float num = red[0] + start_t[tg[0]] + end_t[tg[SEQ - 1]];
    float m = -3.0e38f;
    for (int k = 0; k < NTAG; ++k) m = fmaxf(m, cur[k] + end_t[k]);
    float sum = 0.0f;
    for (int k = 0; k < NTAG; ++k) sum += __expf(cur[k] + end_t[k] - m);
    llh[b] = num - (m + __logf(sum));
  }
}

__global__ void final_mean_kernel(const float* __restrict__ llh, float* __restrict__ out) {
  const int tid = threadIdx.x;                         // 64 threads
  __shared__ float red[64];
  red[tid] = llh[tid];
  __syncthreads();
  for (int off = 32; off > 0; off >>= 1) {
    if (tid < off) red[tid] += red[tid + off];
    __syncthreads();
  }
  if (tid == 0) out[0] = -red[0] / (float)BATCH;
}

// ---------------- launch ----------------
extern "C" void kernel_launch(void* const* d_in, const int* in_sizes, int n_in,
                              void* d_out, int out_size, void* d_ws, size_t ws_size,
                              hipStream_t stream) {
  (void)in_sizes; (void)n_in; (void)out_size; (void)ws_size;
  const int*   batch  = (const int*)d_in[0];
  const int*   tags   = (const int*)d_in[1];
  /* d_in[2] = seq_lengths (all == S, unused) */
  const float* emb    = (const float*)d_in[3];
  const float* w_ih_f = (const float*)d_in[4];
  const float* w_hh_f = (const float*)d_in[5];
  const float* b_f    = (const float*)d_in[6];
  const float* w_ih_b = (const float*)d_in[7];
  const float* w_hh_b = (const float*)d_in[8];
  const float* b_b    = (const float*)d_in[9];
  const float* w_out  = (const float*)d_in[10];
  const float* b_out  = (const float*)d_in[11];
  const float* startt = (const float*)d_in[12];
  const float* endt   = (const float*)d_in[13];
  const float* trans  = (const float*)d_in[14];

  char* ws = (char*)d_ws;
  size_t off = 0;
  auto carve = [&](size_t bytes) { void* p = ws + off; off += (bytes + 255) & ~(size_t)255; return p; };
  bf16_t* X      = (bf16_t*)carve((size_t)MROWS * KPAD * 2);        // 10.5 MB
  bf16_t* wihB   = (bf16_t*)carve((size_t)2 * 10 * 64 * 512 * 2);   // 1.3 MB
  bf16_t* whhB   = (bf16_t*)carve((size_t)2 * 8 * 64 * 512 * 2);    // 1.0 MB
  bf16_t* woutB  = (bf16_t*)carve((size_t)16 * 3 * 512 * 2);        // 48 KB
  float*  xg     = (float*)carve((size_t)2 * MROWS * G4H * 4);      // 134 MB
  float*  hidden = (float*)carve((size_t)MROWS * 2 * HID * 4);      // 33.5 MB
  float*  emis   = (float*)carve((size_t)BATCH * SEQ * NTAG * 4);   // 2.2 MB
  float*  llh    = (float*)carve((size_t)BATCH * 4);

  pack_x_kernel   <<<(MROWS * KPAD) / 256, 256, 0, stream>>>(batch, emb, X);
  pack_wih_kernel <<<(2 * 10 * 64 * 512) / 256, 256, 0, stream>>>(w_ih_f, w_ih_b, wihB);
  pack_whh_kernel <<<(2 * 8 * 64 * 512) / 256, 256, 0, stream>>>(w_hh_f, w_hh_b, whhB);
  pack_wout_kernel<<<(16 * 3 * 512) / 256, 256, 0, stream>>>(w_out, woutB);

  xproj_gemm_kernel<<<(2 * 1024 * 64) / 8, 256, 0, stream>>>(X, wihB, b_f, b_b, xg);
  lstm_scan_kernel <<<2, 1024, 0, stream>>>(xg, whhB, hidden);
  outproj_kernel   <<<(1024 * 3) / 8, 256, 0, stream>>>(hidden, woutB, b_out, emis);
  crf_kernel       <<<BATCH, 64, 0, stream>>>(emis, tags, startt, endt, trans, llh);
  final_mean_kernel<<<1, 64, 0, stream>>>(llh, (float*)d_out);
}
